// GNN_30382598652486
// MI455X (gfx1250) — matrix operand order repaired and checked by reading, not compiled, not run on hardware
//
#include <hip/hip_runtime.h>

typedef __bf16 bf16;
typedef __attribute__((ext_vector_type(16))) __bf16 v16bf;
typedef __attribute__((ext_vector_type(8)))  float  v8f;
typedef __attribute__((ext_vector_type(4)))  unsigned int u32x4;

#define NT 4  // 16-wide output tiles per wave (64 output columns)

// Load a 16-element bf16 fragment as two 16-byte chunks (matches the
// half-wave K-split layout of 16-bit WMMA A/B fragments on gfx1250).
__device__ __forceinline__ v16bf load_frag_pair(const bf16* p_lo, const bf16* p_hi) {
  union { u32x4 u[2]; v16bf v; } t;
  t.u[0] = *reinterpret_cast<const u32x4*>(p_lo);
  t.u[1] = *reinterpret_cast<const u32x4*>(p_hi);
  return t.v;
}

// ---------------------------------------------------------------------------
// Build the two 4x4 node-mixing matrices from edge_index (shared by batch).
// mats[0..15]  = GCN  D^-1/2 (A+I) D^-1/2   (row=dst, col=src)
// mats[16..31] = SAGE mean-aggregation matrix
// ---------------------------------------------------------------------------
__global__ void prep_mats_kernel(const int* __restrict__ ei, float* __restrict__ mats) {
  if (threadIdx.x != 0 || blockIdx.x != 0) return;
  const int E = 12;
  float deg[4] = {0, 0, 0, 0}, cnt[4] = {0, 0, 0, 0};
  for (int e = 0; e < E; ++e) { int d = ei[E + e]; deg[d] += 1.f; cnt[d] += 1.f; }
  for (int i = 0; i < 4; ++i) deg[i] += 1.f;  // self loops
  float dinv[4];
  for (int i = 0; i < 4; ++i) dinv[i] = rsqrtf(deg[i]);
  float g[16], m[16];
  for (int i = 0; i < 16; ++i) { g[i] = 0.f; m[i] = 0.f; }
  for (int e = 0; e < E; ++e) {
    int s = ei[e], d = ei[E + e];
    g[d * 4 + s] += dinv[s] * dinv[d];
    m[d * 4 + s] += 1.f;
  }
  for (int i = 0; i < 4; ++i) g[i * 4 + i] += dinv[i] * dinv[i];
  for (int i = 0; i < 4; ++i) {
    float c = fmaxf(cnt[i], 1.f);
    for (int j = 0; j < 4; ++j) mats[16 + i * 4 + j] = m[i * 4 + j] / c;
  }
  for (int i = 0; i < 16; ++i) mats[i] = g[i];
}

// f32 -> bf16 convert (activations)
__global__ void cvt_bf16_kernel(const float* __restrict__ in, bf16* __restrict__ out, long n) {
  long i = (long)blockIdx.x * blockDim.x + threadIdx.x;
  if (i < n) out[i] = (bf16)in[i];
}

// W [K x N] f32 row-major -> Wt [N x K] bf16 (so B-fragments are contiguous)
__global__ void tconv_kernel(const float* __restrict__ W, bf16* __restrict__ Wt, int K, int N) {
  long i = (long)blockIdx.x * blockDim.x + threadIdx.x;
  long total = (long)K * N;
  if (i >= total) return;
  int k = (int)(i / N), n = (int)(i - (long)k * N);
  Wt[(long)n * K + k] = (bf16)W[i];
}

// ---------------------------------------------------------------------------
// GCN layer: out = relu( Mix_gcn(X @ W) + b ), fused mixing in epilogue.
// X [M x K] bf16, Wt [N x K] bf16, out [M x N] bf16.  One wave = 16x64 tile.
// ---------------------------------------------------------------------------
__global__ void __launch_bounds__(256)
gcn_gemm_kernel(const bf16* __restrict__ X, const bf16* __restrict__ Wt,
                const float* __restrict__ bias, const float* __restrict__ mats,
                bf16* __restrict__ out, int M, int K, int N) {
  const int wave = blockIdx.x * (blockDim.x >> 5) + (threadIdx.x >> 5);
  const int lane = threadIdx.x & 31;
  const int ngrp = N / (16 * NT);
  const int mt = wave / ngrp, ng = wave - mt * ngrp;
  const int m0 = mt * 16, n0 = ng * (16 * NT);
  if (m0 >= M) return;
  const int half = lane >> 4, r = lane & 15;

  v8f acc[NT] = {};
  const bf16* aptr = X + (size_t)(m0 + r) * K + half * 8;
  for (int k0 = 0; k0 < K; k0 += 32) {
    v16bf a = load_frag_pair(aptr + k0, aptr + k0 + 16);
#pragma unroll
    for (int t = 0; t < NT; ++t) {
      const bf16* bptr = Wt + (size_t)(n0 + t * 16 + r) * K + k0 + half * 8;
      v16bf b = load_frag_pair(bptr, bptr + 16);
      acc[t] = __builtin_amdgcn_wmma_f32_16x16x32_bf16(
          false, a, false, b, (short)0, acc[t], false, false);
    }
  }
  float G[16];
#pragma unroll
  for (int i = 0; i < 16; ++i) G[i] = mats[i];
#pragma unroll
  for (int t = 0; t < NT; ++t) {
    const int col = n0 + t * 16 + r;
    const float bv = bias[col];
    float mx[8];
#pragma unroll
    for (int g = 0; g < 8; g += 4)
#pragma unroll
      for (int i = 0; i < 4; ++i)
        mx[g + i] = G[i * 4 + 0] * acc[t][g + 0] + G[i * 4 + 1] * acc[t][g + 1] +
                    G[i * 4 + 2] * acc[t][g + 2] + G[i * 4 + 3] * acc[t][g + 3];
    bf16* orow = out + (size_t)(m0 + half * 8) * N + col;
#pragma unroll
    for (int rr = 0; rr < 8; ++rr) {
      float v = mx[rr] + bv;
      orow[(size_t)rr * N] = (bf16)(v > 0.f ? v : 0.f);
    }
  }
}

// ---------------------------------------------------------------------------
// SAGE layer pre-norm: out = Mix_mean(X @ Wl) + X @ Wr + bl  (no activation,
// L2-normalize+relu done by norm_relu_kernel). Two accumulators, shared A.
// ---------------------------------------------------------------------------
__global__ void __launch_bounds__(256)
sage_gemm_kernel(const bf16* __restrict__ X, const bf16* __restrict__ Wlt,
                 const bf16* __restrict__ Wrt, const float* __restrict__ bias,
                 const float* __restrict__ matm, bf16* __restrict__ out,
                 int M, int K, int N) {
  const int wave = blockIdx.x * (blockDim.x >> 5) + (threadIdx.x >> 5);
  const int lane = threadIdx.x & 31;
  const int ngrp = N / (16 * NT);
  const int mt = wave / ngrp, ng = wave - mt * ngrp;
  const int m0 = mt * 16, n0 = ng * (16 * NT);
  if (m0 >= M) return;
  const int half = lane >> 4, r = lane & 15;

  v8f accl[NT] = {}, accr[NT] = {};
  const bf16* aptr = X + (size_t)(m0 + r) * K + half * 8;
  for (int k0 = 0; k0 < K; k0 += 32) {
    v16bf a = load_frag_pair(aptr + k0, aptr + k0 + 16);
#pragma unroll
    for (int t = 0; t < NT; ++t) {
      const size_t coff = (size_t)(n0 + t * 16 + r) * K + k0 + half * 8;
      v16bf bl = load_frag_pair(Wlt + coff, Wlt + coff + 16);
      accl[t] = __builtin_amdgcn_wmma_f32_16x16x32_bf16(
          false, a, false, bl, (short)0, accl[t], false, false);
      v16bf br = load_frag_pair(Wrt + coff, Wrt + coff + 16);
      accr[t] = __builtin_amdgcn_wmma_f32_16x16x32_bf16(
          false, a, false, br, (short)0, accr[t], false, false);
    }
  }
  float G[16];
#pragma unroll
  for (int i = 0; i < 16; ++i) G[i] = matm[i];
#pragma unroll
  for (int t = 0; t < NT; ++t) {
    const int col = n0 + t * 16 + r;
    const float bv = bias[col];
    float mx[8];
#pragma unroll
    for (int g = 0; g < 8; g += 4)
#pragma unroll
      for (int i = 0; i < 4; ++i)
        mx[g + i] = G[i * 4 + 0] * accl[t][g + 0] + G[i * 4 + 1] * accl[t][g + 1] +
                    G[i * 4 + 2] * accl[t][g + 2] + G[i * 4 + 3] * accl[t][g + 3];
    bf16* orow = out + (size_t)(m0 + half * 8) * N + col;
#pragma unroll
    for (int rr = 0; rr < 8; ++rr)
      orow[(size_t)rr * N] = (bf16)(mx[rr] + accr[t][rr] + bv);
  }
}

// In-place row L2-normalize + relu. One wave32 per row.
__global__ void __launch_bounds__(256)
norm_relu_kernel(bf16* __restrict__ h, int M, int N) {
  const int wave = blockIdx.x * (blockDim.x >> 5) + (threadIdx.x >> 5);
  const int lane = threadIdx.x & 31;
  if (wave >= M) return;
  bf16* row = h + (size_t)wave * N;
  float ss = 0.f;
  for (int c = lane; c < N; c += 32) { float v = (float)row[c]; ss += v * v; }
#pragma unroll
  for (int off = 16; off > 0; off >>= 1) ss += __shfl_xor(ss, off, 32);
  float inv = 1.f / fmaxf(sqrtf(ss), 1e-12f);
  for (int c = lane; c < N; c += 32) {
    float v = (float)row[c] * inv;
    row[c] = (bf16)(v > 0.f ? v : 0.f);
  }
}

// Final FC (K=1024 -> C=10) + softmax. One wave per batch row, Wfc in LDS.
__global__ void __launch_bounds__(256)
fc_softmax_kernel(const bf16* __restrict__ H, const float* __restrict__ Wfc,
                  const float* __restrict__ bfc, float* __restrict__ out,
                  int B, int K, int C) {
  __shared__ float sW[1024 * 10 + 16];
  for (int i = threadIdx.x; i < K * C; i += blockDim.x) sW[i] = Wfc[i];
  for (int i = threadIdx.x; i < C; i += blockDim.x) sW[K * C + i] = bfc[i];
  __syncthreads();
  const int lane = threadIdx.x & 31;
  const int wib = threadIdx.x >> 5;
  const int nwaves = gridDim.x * (blockDim.x >> 5);
  for (int b = blockIdx.x * (blockDim.x >> 5) + wib; b < B; b += nwaves) {
    const bf16* hr = H + (size_t)b * K;
    float acc[10];
#pragma unroll
    for (int c = 0; c < 10; ++c) acc[c] = 0.f;
    for (int k = lane; k < K; k += 32) {
      float a = (float)hr[k];
      const float* w = sW + k * 10;
#pragma unroll
      for (int c = 0; c < 10; ++c) acc[c] += a * w[c];
    }
#pragma unroll
    for (int off = 16; off > 0; off >>= 1)
#pragma unroll
      for (int c = 0; c < 10; ++c) acc[c] += __shfl_xor(acc[c], off, 32);
    float mx = -3.0e38f;
#pragma unroll
    for (int c = 0; c < 10; ++c) { acc[c] += sW[K * 10 + c]; mx = fmaxf(mx, acc[c]); }
    float den = 0.f;
#pragma unroll
    for (int c = 0; c < 10; ++c) { acc[c] = __expf(acc[c] - mx); den += acc[c]; }
    float rinv = 1.f / den;
    if (lane < 10) out[(size_t)b * 10 + lane] = acc[lane] * rinv;
  }
}

extern "C" void kernel_launch(void* const* d_in, const int* in_sizes, int n_in,
                              void* d_out, int out_size, void* d_ws, size_t ws_size,
                              hipStream_t stream) {
  const int B = 32768, Nn = 4, M = B * Nn;  // 131072 rows

  const float* x   = (const float*)d_in[0];
  const int*   ei  = (const int*)d_in[1];
  const float* W1  = (const float*)d_in[2];  const float* b1  = (const float*)d_in[3];
  const float* Wl2 = (const float*)d_in[4];  const float* bl2 = (const float*)d_in[5];
  const float* Wr2 = (const float*)d_in[6];
  const float* Wl3 = (const float*)d_in[7];  const float* bl3 = (const float*)d_in[8];
  const float* Wr3 = (const float*)d_in[9];
  const float* Wl4 = (const float*)d_in[10]; const float* bl4 = (const float*)d_in[11];
  const float* Wr4 = (const float*)d_in[12];
  const float* Wfc = (const float*)d_in[13]; const float* bfc = (const float*)d_in[14];

  char* ws = (char*)d_ws;
  size_t off = 0;
  auto alloc = [&](size_t bytes) { char* p = ws + off; off += (bytes + 255) & ~(size_t)255; return p; };

  float* mats = (float*)alloc(32 * sizeof(float));
  bf16* W1t  = (bf16*)alloc((size_t)1024 * 64 * 2);
  bf16* Wl2t = (bf16*)alloc((size_t)512 * 1024 * 2);
  bf16* Wr2t = (bf16*)alloc((size_t)512 * 1024 * 2);
  bf16* Wl3t = (bf16*)alloc((size_t)256 * 512 * 2);
  bf16* Wr3t = (bf16*)alloc((size_t)256 * 512 * 2);
  bf16* Wl4t = (bf16*)alloc((size_t)256 * 256 * 2);
  bf16* Wr4t = (bf16*)alloc((size_t)256 * 256 * 2);
  bf16* X16  = (bf16*)alloc((size_t)M * 64 * 2);
  bf16* H1   = (bf16*)alloc((size_t)M * 1024 * 2);
  bf16* H2   = (bf16*)alloc((size_t)M * 512 * 2);
  bf16* H3   = (bf16*)(H1);                           // H1 dead after layer 2
  bf16* H4   = (bf16*)((char*)H1 + (size_t)M * 256 * 2);

  prep_mats_kernel<<<1, 32, 0, stream>>>(ei, mats);

  auto tgrid = [](long n) { return (unsigned)((n + 255) / 256); };
  tconv_kernel<<<tgrid(64L * 1024), 256, 0, stream>>>(W1, W1t, 64, 1024);
  tconv_kernel<<<tgrid(1024L * 512), 256, 0, stream>>>(Wl2, Wl2t, 1024, 512);
  tconv_kernel<<<tgrid(1024L * 512), 256, 0, stream>>>(Wr2, Wr2t, 1024, 512);
  tconv_kernel<<<tgrid(512L * 256), 256, 0, stream>>>(Wl3, Wl3t, 512, 256);
  tconv_kernel<<<tgrid(512L * 256), 256, 0, stream>>>(Wr3, Wr3t, 512, 256);
  tconv_kernel<<<tgrid(256L * 256), 256, 0, stream>>>(Wl4, Wl4t, 256, 256);
  tconv_kernel<<<tgrid(256L * 256), 256, 0, stream>>>(Wr4, Wr4t, 256, 256);
  cvt_bf16_kernel<<<tgrid((long)M * 64), 256, 0, stream>>>(x, X16, (long)M * 64);

  // Layer 1 (GCN, K=64 -> N=1024): waves = (M/16)*(1024/64) = 131072
  gcn_gemm_kernel<<<131072 / 8, 256, 0, stream>>>(X16, W1t, b1, mats, H1, M, 64, 1024);

  // Layer 2 (SAGE, 1024 -> 512): waves = 8192*8 = 65536
  sage_gemm_kernel<<<65536 / 8, 256, 0, stream>>>(H1, Wl2t, Wr2t, bl2, mats + 16, H2, M, 1024, 512);
  norm_relu_kernel<<<M / 8, 256, 0, stream>>>(H2, M, 512);

  // Layer 3 (SAGE, 512 -> 256): waves = 8192*4 = 32768
  sage_gemm_kernel<<<32768 / 8, 256, 0, stream>>>(H2, Wl3t, Wr3t, bl3, mats + 16, H3, M, 512, 256);
  norm_relu_kernel<<<M / 8, 256, 0, stream>>>(H3, M, 256);

  // Layer 4 (SAGE, 256 -> 256)
  sage_gemm_kernel<<<32768 / 8, 256, 0, stream>>>(H3, Wl4t, Wr4t, bl4, mats + 16, H4, M, 256, 256);
  norm_relu_kernel<<<M / 8, 256, 0, stream>>>(H4, M, 256);

  // FC + softmax: H4 viewed as [B x 1024] (row-major b,node,feat = reshape)
  fc_softmax_kernel<<<512, 256, 0, stream>>>(H4, Wfc, bfc, (float*)d_out, B, 1024, 10);

  (void)in_sizes; (void)n_in; (void)out_size; (void)ws_size;
}
